// EnBaseLayer_40596030882310
// MI455X (gfx1250) — matrix operand order repaired and checked by reading, not compiled, not run on hardware
//
#include <hip/hip_runtime.h>

#define HDIM 64

typedef __attribute__((ext_vector_type(16))) _Float16 v16h;
typedef __attribute__((ext_vector_type(8)))  _Float16 v8h;
typedef __attribute__((ext_vector_type(8)))  float    v8f;

union AFrag { v16h v; v8h h[2]; };

__device__ __forceinline__ float sigmoidf_(float t) {
    return 1.0f / (1.0f + __expf(-t));
}

__device__ __forceinline__ v8f vzero8() {
    v8f r;
#pragma unroll
    for (int i = 0; i < 8; ++i) r[i] = 0.0f;
    return r;
}

__device__ __forceinline__ v8h cvt8_f32_f16(const float* __restrict__ p) {
    v8h r;
#pragma unroll
    for (int i = 0; i < 8; ++i) r[i] = (_Float16)p[i];
    return r;
}

__device__ __forceinline__ void atomic_add_f32(float* p, float v) {
    // relaxed, agent scope -> global_atomic_add_f32 on gfx1250
    (void)__hip_atomic_fetch_add(p, v, __ATOMIC_RELAXED, __HIP_MEMORY_SCOPE_AGENT);
}

// Pre-swizzle a [K x 64] f32 weight matrix into f16 WMMA B-fragments in LDS.
// Fragment f = kk*4 + nt (K-step kk of 32, N-tile nt of 16).
// B layout (16-bit, 32x16): lanes 0-15: N=lane, K = kk*32 + 0..7, 16..23 (2 per VGPR);
//                           lanes 16-31: N=lane-16, K = kk*32 + 8..15, 24..31.
// Stored lane-contiguous: fragment f, lane l -> halves [f*512 + l*16 .. +15].
__device__ __forceinline__ void fill_bfrags(_Float16* lds, const float* __restrict__ W,
                                            int nFrag, int tid) {
    const int total = nFrag * 512;
    for (int e = tid; e < total; e += 256) {
        const int f  = e >> 9;
        const int l  = (e >> 4) & 31;
        const int j  = e & 15;
        const int kk = f >> 2;
        const int nt = f & 3;
        const int K  = kk * 32 + ((l & 16) ? 8 : 0) + (j & 7) + ((j & 8) ? 16 : 0);
        const int n  = nt * 16 + (l & 15);
        lds[e] = (_Float16)W[K * 64 + n];
    }
}

__global__ void prep_kernel(const float* __restrict__ h, const float* __restrict__ x,
                            _Float16* __restrict__ hf, float* __restrict__ mi,
                            float* __restrict__ xout, int nH, int nX) {
    const int i = blockIdx.x * blockDim.x + threadIdx.x;
    if (i < nH) {
        hf[i] = (_Float16)h[i];
        mi[i] = 0.0f;
    }
    if (i < nX) xout[i] = x[i];
}

// One 16-edge tile per wave32.  A = [h[dst] | h[src]] (16x128 f16),
// layer1: A @ We1 (4 K-steps x 4 N-tiles = 16 wmma), relu -> LDS staging (f16),
// layer2: @ We2 (2 x 4 = 8 wmma), relu -> mij; gate = sigmoid((mij@Winf+binf)*edge_dis);
// atomic segment-sum of mij*gate into mi[dst].
__global__ __launch_bounds__(256) void edge_kernel(
    const float* __restrict__ x, const int* __restrict__ ei,
    const float* __restrict__ We1, const float* __restrict__ be1,
    const float* __restrict__ We2, const float* __restrict__ be2,
    const float* __restrict__ Winf, const float* __restrict__ binf,
    const _Float16* __restrict__ hf, float* __restrict__ mi, int nEdges) {
    __shared__ alignas(16) _Float16 ldsW1[16 * 512];
    __shared__ alignas(16) _Float16 ldsW2[8 * 512];
    __shared__ alignas(16) _Float16 stag[8 * 16 * HDIM];

    const int tid = threadIdx.x;
    fill_bfrags(ldsW1, We1, 16, tid);
    fill_bfrags(ldsW2, We2, 8, tid);
    __syncthreads();

    const int lane = tid & 31;
    const int wid  = tid >> 5;
    const int m    = lane & 15;           // my A-row / my N-column-in-tile
    const int hi8  = (lane & 16) ? 8 : 0; // half-wave K/M offset
    _Float16* myStag = stag + wid * (16 * HDIM);

    float b1v[4], b2v[4], wv[4];
#pragma unroll
    for (int nt = 0; nt < 4; ++nt) {
        const int n = nt * 16 + m;
        b1v[nt] = be1[n];
        b2v[nt] = be2[n];
        wv[nt]  = Winf[n];
    }
    const float binf_s = binf[0];

    const int nTiles = nEdges >> 4;
    const int wavesTotal = gridDim.x * 8;
    for (int tile = blockIdx.x * 8 + wid; tile < nTiles; tile += wavesTotal) {
        const int e   = tile * 16 + m;
        const int src = ei[e];
        const int dst = ei[nEdges + e];

        const float dx = x[dst * 3 + 0] - x[src * 3 + 0];
        const float dy = x[dst * 3 + 1] - x[src * 3 + 1];
        const float dz = x[dst * 3 + 2] - x[src * 3 + 2];
        const float edis = sigmoidf_(30.0f / (sqrtf(dx * dx + dy * dy + dz * dz) + 1e-8f));

        const _Float16* hdst = hf + (size_t)dst * HDIM;
        const _Float16* hsrc = hf + (size_t)src * HDIM;

        // A fragments (16-bit A layout: lane m, halves = K klo..klo+7, klo+16..klo+23)
        AFrag a[4];
#pragma unroll
        for (int kk = 0; kk < 4; ++kk) {
            const _Float16* row = (kk < 2) ? hdst : hsrc;
            const int koff = (kk & 1) * 32 + hi8;
            a[kk].h[0] = *(const v8h*)(row + koff);
            a[kk].h[1] = *(const v8h*)(row + koff + 16);
        }

        // layer 1
        v8f c1[4];
#pragma unroll
        for (int nt = 0; nt < 4; ++nt) {
            v8f acc = vzero8();
#pragma unroll
            for (int kk = 0; kk < 4; ++kk) {
                AFrag b;
                const _Float16* bp = ldsW1 + (kk * 4 + nt) * 512 + lane * 16;
                b.h[0] = *(const v8h*)(bp);
                b.h[1] = *(const v8h*)(bp + 8);
                acc = __builtin_amdgcn_wmma_f32_16x16x32_f16(false, a[kk].v, false, b.v,
                                                             (short)0, acc, false, false);
            }
            c1[nt] = acc;
        }

        // bias + relu, stage to LDS row-major f16 [16][64] (C->A transpose bounce)
#pragma unroll
        for (int nt = 0; nt < 4; ++nt) {
            const int n = nt * 16 + m;
#pragma unroll
            for (int r = 0; r < 8; ++r) {
                const float v = fmaxf(c1[nt][r] + b1v[nt], 0.0f);
                myStag[(hi8 + r) * HDIM + n] = (_Float16)v;
            }
        }

        // layer 2 A fragments from staging
        AFrag a2[2];
        {
            const _Float16* row = myStag + m * HDIM;
#pragma unroll
            for (int kk = 0; kk < 2; ++kk) {
                const int koff = kk * 32 + hi8;
                a2[kk].h[0] = *(const v8h*)(row + koff);
                a2[kk].h[1] = *(const v8h*)(row + koff + 16);
            }
        }

        v8f c2[4];
        float eij[8] = {0, 0, 0, 0, 0, 0, 0, 0};
#pragma unroll
        for (int nt = 0; nt < 4; ++nt) {
            v8f acc = vzero8();
#pragma unroll
            for (int kk = 0; kk < 2; ++kk) {
                AFrag b;
                const _Float16* bp = ldsW2 + (kk * 4 + nt) * 512 + lane * 16;
                b.h[0] = *(const v8h*)(bp);
                b.h[1] = *(const v8h*)(bp + 8);
                acc = __builtin_amdgcn_wmma_f32_16x16x32_f16(false, a2[kk].v, false, b.v,
                                                             (short)0, acc, false, false);
            }
#pragma unroll
            for (int r = 0; r < 8; ++r) {
                const float v = fmaxf(acc[r] + b2v[nt], 0.0f); // mij
                acc[r] = v;
                eij[r] += v * wv[nt];
            }
            c2[nt] = acc;
        }

        // reduce eij across the 16-lane N-group; gate; atomic scatter-add
        float wrow[8];
#pragma unroll
        for (int r = 0; r < 8; ++r) {
            float s = eij[r];
            s += __shfl_xor(s, 1);
            s += __shfl_xor(s, 2);
            s += __shfl_xor(s, 4);
            s += __shfl_xor(s, 8);
            s += binf_s;
            const int mr = hi8 + r;
            const float ed = __shfl(edis, mr);
            wrow[r] = sigmoidf_(s * ed);
        }

#pragma unroll
        for (int r = 0; r < 8; ++r) {
            const int mr = hi8 + r;
            const int d  = __shfl(dst, mr);
            float* out = mi + (size_t)d * HDIM;
#pragma unroll
            for (int nt = 0; nt < 4; ++nt) {
                atomic_add_f32(out + nt * 16 + m, c2[nt][r] * wrow[r]);
            }
        }
    }
}

// One 16-node tile per wave32.  A = [mi | h] (16x128), layer1 relu, layer2 (+bias),
// residual + LayerNorm, store h_out.
__global__ __launch_bounds__(256) void node_kernel(
    const float* __restrict__ h, const _Float16* __restrict__ hf,
    const float* __restrict__ mi,
    const float* __restrict__ Wn1, const float* __restrict__ bn1,
    const float* __restrict__ Wn2, const float* __restrict__ bn2,
    const float* __restrict__ ln_g, const float* __restrict__ ln_b,
    float* __restrict__ hout, int nNodes) {
    __shared__ alignas(16) _Float16 ldsW1[16 * 512];
    __shared__ alignas(16) _Float16 ldsW2[8 * 512];
    __shared__ alignas(16) _Float16 stag[8 * 16 * HDIM];

    const int tid = threadIdx.x;
    fill_bfrags(ldsW1, Wn1, 16, tid);
    fill_bfrags(ldsW2, Wn2, 8, tid);
    __syncthreads();

    const int lane = tid & 31;
    const int wid  = tid >> 5;
    const int m    = lane & 15;
    const int hi8  = (lane & 16) ? 8 : 0;
    _Float16* myStag = stag + wid * (16 * HDIM);

    float b1v[4], b2v[4], gv[4], bv[4];
#pragma unroll
    for (int nt = 0; nt < 4; ++nt) {
        const int n = nt * 16 + m;
        b1v[nt] = bn1[n];
        b2v[nt] = bn2[n];
        gv[nt]  = ln_g[n];
        bv[nt]  = ln_b[n];
    }

    const int nTiles = nNodes >> 4;
    const int wavesTotal = gridDim.x * 8;
    for (int tile = blockIdx.x * 8 + wid; tile < nTiles; tile += wavesTotal) {
        const int node = tile * 16 + m;
        const float*    mrow = mi + (size_t)node * HDIM;
        const _Float16* hrow = hf + (size_t)node * HDIM;

        AFrag a[4];
#pragma unroll
        for (int kk = 0; kk < 2; ++kk) { // mi half (f32 -> f16 on the fly)
            const int koff = kk * 32 + hi8;
            a[kk].h[0] = cvt8_f32_f16(mrow + koff);
            a[kk].h[1] = cvt8_f32_f16(mrow + koff + 16);
        }
#pragma unroll
        for (int kk = 2; kk < 4; ++kk) { // h half (already f16)
            const int koff = (kk - 2) * 32 + hi8;
            a[kk].h[0] = *(const v8h*)(hrow + koff);
            a[kk].h[1] = *(const v8h*)(hrow + koff + 16);
        }

        v8f c1[4];
#pragma unroll
        for (int nt = 0; nt < 4; ++nt) {
            v8f acc = vzero8();
#pragma unroll
            for (int kk = 0; kk < 4; ++kk) {
                AFrag b;
                const _Float16* bp = ldsW1 + (kk * 4 + nt) * 512 + lane * 16;
                b.h[0] = *(const v8h*)(bp);
                b.h[1] = *(const v8h*)(bp + 8);
                acc = __builtin_amdgcn_wmma_f32_16x16x32_f16(false, a[kk].v, false, b.v,
                                                             (short)0, acc, false, false);
            }
            c1[nt] = acc;
        }

#pragma unroll
        for (int nt = 0; nt < 4; ++nt) {
            const int n = nt * 16 + m;
#pragma unroll
            for (int r = 0; r < 8; ++r) {
                const float v = fmaxf(c1[nt][r] + b1v[nt], 0.0f);
                myStag[(hi8 + r) * HDIM + n] = (_Float16)v;
            }
        }

        AFrag a2[2];
        {
            const _Float16* row = myStag + m * HDIM;
#pragma unroll
            for (int kk = 0; kk < 2; ++kk) {
                const int koff = kk * 32 + hi8;
                a2[kk].h[0] = *(const v8h*)(row + koff);
                a2[kk].h[1] = *(const v8h*)(row + koff + 16);
            }
        }

        v8f z[4];
        float zs[8]  = {0, 0, 0, 0, 0, 0, 0, 0};
        float z2s[8] = {0, 0, 0, 0, 0, 0, 0, 0};
        const int nodeBase = tile * 16;
#pragma unroll
        for (int nt = 0; nt < 4; ++nt) {
            v8f acc = vzero8();
#pragma unroll
            for (int kk = 0; kk < 2; ++kk) {
                AFrag b;
                const _Float16* bp = ldsW2 + (kk * 4 + nt) * 512 + lane * 16;
                b.h[0] = *(const v8h*)(bp);
                b.h[1] = *(const v8h*)(bp + 8);
                acc = __builtin_amdgcn_wmma_f32_16x16x32_f16(false, a2[kk].v, false, b.v,
                                                             (short)0, acc, false, false);
            }
            const int n = nt * 16 + m;
#pragma unroll
            for (int r = 0; r < 8; ++r) { // residual: z = h + upd (no relu on layer 2)
                const float val = acc[r] + b2v[nt] +
                                  h[(size_t)(nodeBase + hi8 + r) * HDIM + n];
                z[nt][r] = val;
                zs[r]  += val;
                z2s[r] += val * val;
            }
        }

        // LayerNorm over the 64-wide row (spread across 16 lanes x 4 N-tiles)
#pragma unroll
        for (int r = 0; r < 8; ++r) {
            float s = zs[r], s2 = z2s[r];
#pragma unroll
            for (int off = 1; off < 16; off <<= 1) {
                s  += __shfl_xor(s, off);
                s2 += __shfl_xor(s2, off);
            }
            const float mean = s * (1.0f / 64.0f);
            const float var  = s2 * (1.0f / 64.0f) - mean * mean;
            const float rstd = rsqrtf(var + 1e-5f);
            const size_t rowoff = (size_t)(nodeBase + hi8 + r) * HDIM;
#pragma unroll
            for (int nt = 0; nt < 4; ++nt) {
                hout[rowoff + nt * 16 + m] = (z[nt][r] - mean) * rstd * gv[nt] + bv[nt];
            }
        }
    }
}

extern "C" void kernel_launch(void* const* d_in, const int* in_sizes, int n_in,
                              void* d_out, int out_size, void* d_ws, size_t ws_size,
                              hipStream_t stream) {
    const float* h    = (const float*)d_in[0];
    const float* x    = (const float*)d_in[1];
    const int*   ei   = (const int*)d_in[2];
    const float* We1  = (const float*)d_in[3];
    const float* be1  = (const float*)d_in[4];
    const float* We2  = (const float*)d_in[5];
    const float* be2  = (const float*)d_in[6];
    const float* Winf = (const float*)d_in[7];
    const float* binf = (const float*)d_in[8];
    const float* Wn1  = (const float*)d_in[9];
    const float* bn1  = (const float*)d_in[10];
    const float* Wn2  = (const float*)d_in[11];
    const float* bn2  = (const float*)d_in[12];
    const float* ln_g = (const float*)d_in[13];
    const float* ln_b = (const float*)d_in[14];

    const int N = in_sizes[0] / HDIM;
    const int E = in_sizes[2] / 2;

    _Float16* hf = (_Float16*)d_ws;                                          // N*64 f16
    float* mi = (float*)((char*)d_ws + (size_t)N * HDIM * sizeof(_Float16)); // N*64 f32

    float* hout = (float*)d_out;
    float* xout = hout + (size_t)N * HDIM;

    const int nH = N * HDIM;
    const int nX = N * 3;

    prep_kernel<<<(nH + 255) / 256, 256, 0, stream>>>(h, x, hf, mi, xout, nH, nX);
    edge_kernel<<<1024, 256, 0, stream>>>(x, ei, We1, be1, We2, be2, Winf, binf, hf, mi, E);
    node_kernel<<<512, 256, 0, stream>>>(h, hf, mi, Wn1, bn1, Wn2, bn2, ln_g, ln_b, hout, N);
}
